// EGRETLayer_47502338294070
// MI455X (gfx1250) — matrix operand order repaired and checked by reading, not compiled, not run on hardware
//
// EGRET layer for MI455X (gfx1250). bf16 WMMA (v_wmma_f32_16x16x32_bf16) for all
// GEMMs, fp32 accumulation. Per-edge MLPs fused into one wave-tiled kernel:
// each wave owns 16 edges, assembles the [16 x 224] bf16 edge-input tile in LDS,
// and drives the WMMA chain (attention 224->64, message 224->128->64, coord 64->64).
// Global softmax via deterministic tree reductions; segment-sum via f32 atomics.

#include <hip/hip_runtime.h>
#include <hip/hip_bf16.h>

typedef __attribute__((ext_vector_type(16))) __bf16 v16bf;
typedef __attribute__((ext_vector_type(8)))  __bf16 v8bf;
typedef __attribute__((ext_vector_type(8)))  float  v8f;

#define EK_WAVES 4

__device__ __forceinline__ v16bf bf_cat(v8bf lo, v8bf hi) {
  return __builtin_shufflevector(lo, hi, 0,1,2,3,4,5,6,7,8,9,10,11,12,13,14,15);
}
// B fragment: weights pre-transposed to Wt[n][k]; lane n = lane&15 reads 16
// contiguous bf16 at K = kbase + 16*(lane>=16)  (contiguous k order, same as load_afrag).
__device__ __forceinline__ v16bf load_bfrag(const __bf16* p) {
  v8bf lo = *(const v8bf*)p;
  v8bf hi = *(const v8bf*)(p + 8);
  return bf_cat(lo, hi);
}
// A fragment (16-bit A 16x32): lanes 0-15: K = kb+0..15;
// lanes 16-31: K = kb+16..31.  Two 16B chunks per lane.
__device__ __forceinline__ v16bf load_afrag(const __bf16* row, int kbase, int hf) {
  const __bf16* p = row + kbase + hf * 16;
  v8bf lo = *(const v8bf*)p;
  v8bf hi = *(const v8bf*)(p + 8);
  return bf_cat(lo, hi);
}
__device__ __forceinline__ v16bf load_afrag_f32(const float* row, int kbase, int hf) {
  const float* p = row + kbase + hf * 16;
  v16bf a;
#pragma unroll
  for (int i = 0; i < 8; ++i) a[i] = (__bf16)p[i];
#pragma unroll
  for (int i = 0; i < 8; ++i) a[8 + i] = (__bf16)p[8 + i];
  return a;
}
__device__ __forceinline__ v8f wmma_bf16(v16bf a, v16bf b, v8f c) {
  return __builtin_amdgcn_wmma_f32_16x16x32_bf16(false, a, false, b, (short)0, c,
                                                 false, false);
}

// ---------------- utility kernels ----------------
__global__ void fill_zero_kernel(float* __restrict__ p, long n) {
  long i = (long)blockIdx.x * blockDim.x + threadIdx.x;
  if (i < n) p[i] = 0.f;
}

// dst[n*K + k] = bf16(src[k*Nn + n])   (row-major [K][Nn] -> transposed bf16)
__global__ void transpose_bf16_kernel(const float* __restrict__ src,
                                      __bf16* __restrict__ dst, int K, int Nn) {
  int i = blockIdx.x * blockDim.x + threadIdx.x;
  if (i >= K * Nn) return;
  int n = i / K, k = i % K;
  dst[i] = (__bf16)src[k * Nn + n];
}

// ---------------- node transform: h = node_feats @ Wn + bn ----------------
__global__ __launch_bounds__(256) void node_kernel(
    const float* __restrict__ nf, const __bf16* __restrict__ WnT,
    const float* __restrict__ bn, float* __restrict__ hOut,
    __bf16* __restrict__ hbOut, int Nn) {
  const int w = threadIdx.x >> 5, lane = threadIdx.x & 31;
  const int el = lane & 15, hf = lane >> 4;
  const int tile = blockIdx.x * 8 + w;
  if (tile * 16 >= Nn) return;  // wave-uniform skip; active waves keep EXEC all-ones
  int ar = tile * 16 + el; if (ar >= Nn) ar = Nn - 1;
  const float* arow = nf + (size_t)ar * 64;
  v8f c[4];
#pragma unroll
  for (int nt = 0; nt < 4; ++nt) {
    float b = bn[nt * 16 + el];
#pragma unroll
    for (int j = 0; j < 8; ++j) c[nt][j] = b;
  }
#pragma unroll
  for (int kt = 0; kt < 2; ++kt) {
    v16bf a = load_afrag_f32(arow, kt * 32, hf);
#pragma unroll
    for (int nt = 0; nt < 4; ++nt) {
      v16bf b = load_bfrag(WnT + (size_t)(nt * 16 + el) * 64 + kt * 32 + hf * 16);
      c[nt] = wmma_bf16(a, b, c[nt]);
    }
  }
#pragma unroll
  for (int nt = 0; nt < 4; ++nt) {
#pragma unroll
    for (int j = 0; j < 8; ++j) {
      int m = tile * 16 + j + 8 * hf;
      if (m < Nn) {
        size_t idx = (size_t)m * 64 + nt * 16 + el;
        float v = c[nt][j];
        hOut[idx]  = v;
        hbOut[idx] = (__bf16)v;
      }
    }
  }
}

// ---------------- fused per-edge kernel ----------------
__global__ __launch_bounds__(128) void edge_kernel(
    const float* __restrict__ edge_feats, const float* __restrict__ coords,
    const int* __restrict__ srcI, const int* __restrict__ dstI,
    const float* __restrict__ cm_w1, const float* __restrict__ cm_b1,
    const float* __restrict__ cm_w2, const float* __restrict__ cm_b2,
    const float* __restrict__ de_w1, const float* __restrict__ de_b1,
    const float* __restrict__ de_w2, const float* __restrict__ de_b2,
    const __bf16* __restrict__ hb,
    const __bf16* __restrict__ am_w1T, const float* __restrict__ am_b1,
    const float* __restrict__ am_w2v, const float* __restrict__ am_b2,
    const __bf16* __restrict__ mm_w1T, const float* __restrict__ mm_b1,
    const __bf16* __restrict__ mm_w2T, const float* __restrict__ mm_b2,
    const __bf16* __restrict__ cu_w1T, const float* __restrict__ cu_b1,
    const float* __restrict__ cu_w2v, const float* __restrict__ cu_b2,
    float* __restrict__ logits, float* __restrict__ msg,
    float* __restrict__ cwraw, float* __restrict__ relunit, int E_) {
  __shared__ __align__(16) __bf16 Atile[EK_WAVES][16 * 224];  // 28 KB
  __shared__ __align__(16) float  scr  [EK_WAVES][16 * 64];   // 16 KB
  __shared__ __align__(16) float  cft  [EK_WAVES][16 * 64];   // 16 KB

  const int w = threadIdx.x >> 5;
  const int lane = threadIdx.x & 31;
  const int el = lane & 15;
  const int hf = lane >> 4;

  const long e_base = ((long)blockIdx.x * EK_WAVES + w) * 16;
  const long e = e_base + el;
  const long ecl = (e < E_) ? e : (long)E_ - 1;   // clamp reads; writes guarded
  const bool ev = (e < E_);

  // ---- Phase 0: per-edge scalars + assemble [16 x 224] bf16 A-tile in LDS ----
  const int s = srcI[ecl];
  const int d = dstI[ecl];
  const float r0 = coords[d * 3 + 0] - coords[s * 3 + 0];
  const float r1 = coords[d * 3 + 1] - coords[s * 3 + 1];
  const float r2 = coords[d * 3 + 2] - coords[s * 3 + 2];
  const float dist = sqrtf(r0 * r0 + r1 * r1 + r2 * r2);
  const float dinv = 1.0f / fmaxf(dist, 1e-12f);
  if (ev && hf == 0) {
    relunit[ecl * 3 + 0] = r0 * dinv;
    relunit[ecl * 3 + 1] = r1 * dinv;
    relunit[ecl * 3 + 2] = r2 * dinv;
  }
  {  // distance_expansion MLP 1->16->64 -> A cols [128,192)
    float hd[16];
#pragma unroll
    for (int k = 0; k < 16; ++k) hd[k] = fmaxf(fmaf(dist, de_w1[k], de_b1[k]), 0.f);
    for (int jj = 0; jj < 32; ++jj) {
      int j = hf * 32 + jj;
      float acc = de_b2[j];
#pragma unroll
      for (int k = 0; k < 16; ++k) acc = fmaf(hd[k], de_w2[k * 64 + j], acc);
      Atile[w][el * 224 + 128 + j] = (__bf16)acc;
    }
  }
  {  // coord MLP 3->16->64 -> cft (fp32, added to messages later)
    float hc[16];
#pragma unroll
    for (int k = 0; k < 16; ++k)
      hc[k] = fmaxf(fmaf(r2, cm_w1[32 + k],
               fmaf(r1, cm_w1[16 + k], fmaf(r0, cm_w1[k], cm_b1[k]))), 0.f);
    for (int jj = 0; jj < 32; ++jj) {
      int j = hf * 32 + jj;
      float acc = cm_b2[j];
#pragma unroll
      for (int k = 0; k < 16; ++k) acc = fmaf(hc[k], cm_w2[k * 64 + j], acc);
      cft[w][el * 64 + j] = acc;
    }
  }
  {  // h[src] -> cols [0,64) (half 0), h[dst] -> cols [64,128) (half 1)
    const __bf16* hrow = hb + (size_t)(hf ? d : s) * 64;
#pragma unroll
    for (int f = 0; f < 64; f += 8)
      *(v8bf*)&Atile[w][el * 224 + hf * 64 + f] = *(const v8bf*)(hrow + f);
  }
  {  // edge feats bf16 -> cols [192,224), 16 per half
    const float* er = edge_feats + (size_t)ecl * 32 + hf * 16;
#pragma unroll
    for (int t = 0; t < 16; ++t)
      Atile[w][el * 224 + 192 + hf * 16 + t] = (__bf16)er[t];
  }
  __syncthreads();

  // ---- Phase 1: stage-1 GEMMs (shared A frags): attn 224->64, msg 224->128 ----
  v8f cam[4], cmm[8];
#pragma unroll
  for (int nt = 0; nt < 4; ++nt) {
    float b = am_b1[nt * 16 + el];
#pragma unroll
    for (int j = 0; j < 8; ++j) cam[nt][j] = b;
  }
#pragma unroll
  for (int nt = 0; nt < 8; ++nt) {
    float b = mm_b1[nt * 16 + el];
#pragma unroll
    for (int j = 0; j < 8; ++j) cmm[nt][j] = b;
  }
  {
    const __bf16* arow = &Atile[w][el * 224];
#pragma unroll
    for (int kt = 0; kt < 7; ++kt) {
      v16bf a = load_afrag(arow, kt * 32, hf);
#pragma unroll
      for (int nt = 0; nt < 4; ++nt) {
        v16bf b = load_bfrag(am_w1T + (size_t)(nt * 16 + el) * 224 + kt * 32 + hf * 16);
        cam[nt] = wmma_bf16(a, b, cam[nt]);
      }
#pragma unroll
      for (int nt = 0; nt < 8; ++nt) {
        v16bf b = load_bfrag(mm_w1T + (size_t)(nt * 16 + el) * 224 + kt * 32 + hf * 16);
        cmm[nt] = wmma_bf16(a, b, cmm[nt]);
      }
    }
  }

  // ---- Phase 2: attention logit = relu(hidden) . am_w2 + am_b2 ----
#pragma unroll
  for (int nt = 0; nt < 4; ++nt) {
#pragma unroll
    for (int j = 0; j < 8; ++j)
      scr[w][(j + 8 * hf) * 64 + nt * 16 + el] = fmaxf(cam[nt][j], 0.f);
  }
  __syncthreads();
  {
    float p = 0.f;
    for (int kk = 0; kk < 32; ++kk) {
      int k = hf * 32 + kk;
      p = fmaf(scr[w][el * 64 + k], am_w2v[k], p);
    }
    p += __shfl_xor(p, 16);
    if (ev && hf == 0) logits[ecl] = p + am_b2[0];
  }
  __syncthreads();

  // ---- Phase 3: relu(msg hidden) -> reuse A-tile as [16 x 128] bf16 ----
#pragma unroll
  for (int nt = 0; nt < 8; ++nt) {
#pragma unroll
    for (int j = 0; j < 8; ++j)
      Atile[w][(j + 8 * hf) * 128 + nt * 16 + el] = (__bf16)fmaxf(cmm[nt][j], 0.f);
  }
  __syncthreads();

  // ---- Phase 4: stage-2 GEMM 128->64, add coord_feats -> messages ----
  v8f c2[4];
#pragma unroll
  for (int nt = 0; nt < 4; ++nt) {
    float b = mm_b2[nt * 16 + el];
#pragma unroll
    for (int j = 0; j < 8; ++j) c2[nt][j] = b;
  }
  {
    const __bf16* arow = &Atile[w][el * 128];
#pragma unroll
    for (int kt = 0; kt < 4; ++kt) {
      v16bf a = load_afrag(arow, kt * 32, hf);
#pragma unroll
      for (int nt = 0; nt < 4; ++nt) {
        v16bf b = load_bfrag(mm_w2T + (size_t)(nt * 16 + el) * 128 + kt * 32 + hf * 16);
        c2[nt] = wmma_bf16(a, b, c2[nt]);
      }
    }
  }
#pragma unroll
  for (int nt = 0; nt < 4; ++nt) {
#pragma unroll
    for (int j = 0; j < 8; ++j) {
      int m = j + 8 * hf;
      int col = nt * 16 + el;
      float v = c2[nt][j] + cft[w][m * 64 + col];
      c2[nt][j] = v;
      long ge = e_base + m;
      if (ge < E_) msg[(size_t)ge * 64 + col] = v;
    }
  }
  __syncthreads();

  // ---- Phase 5: coord weight = relu(messages @ cu_w1 + b) . cu_w2 + b2 ----
#pragma unroll
  for (int nt = 0; nt < 4; ++nt) {
#pragma unroll
    for (int j = 0; j < 8; ++j)
      Atile[w][(j + 8 * hf) * 64 + nt * 16 + el] = (__bf16)c2[nt][j];
  }
  __syncthreads();
  v8f c3[4];
#pragma unroll
  for (int nt = 0; nt < 4; ++nt) {
    float b = cu_b1[nt * 16 + el];
#pragma unroll
    for (int j = 0; j < 8; ++j) c3[nt][j] = b;
  }
  {
    const __bf16* arow = &Atile[w][el * 64];
#pragma unroll
    for (int kt = 0; kt < 2; ++kt) {
      v16bf a = load_afrag(arow, kt * 32, hf);
#pragma unroll
      for (int nt = 0; nt < 4; ++nt) {
        v16bf b = load_bfrag(cu_w1T + (size_t)(nt * 16 + el) * 64 + kt * 32 + hf * 16);
        c3[nt] = wmma_bf16(a, b, c3[nt]);
      }
    }
  }
#pragma unroll
  for (int nt = 0; nt < 4; ++nt) {
#pragma unroll
    for (int j = 0; j < 8; ++j)
      scr[w][(j + 8 * hf) * 64 + nt * 16 + el] = fmaxf(c3[nt][j], 0.f);
  }
  __syncthreads();
  {
    float q = 0.f;
    for (int kk = 0; kk < 32; ++kk) {
      int k = hf * 32 + kk;
      q = fmaf(scr[w][el * 64 + k], cu_w2v[k], q);
    }
    q += __shfl_xor(q, 16);
    if (ev && hf == 0) cwraw[ecl] = q + cu_b2[0];
  }
}

// ---------------- global softmax reductions (deterministic trees) ----------------
__global__ void reduce_max_partial(const float* __restrict__ x, int n,
                                   float* __restrict__ part) {
  __shared__ float sm[256];
  float m = -3.0e38f;
  for (int i = blockIdx.x * blockDim.x + threadIdx.x; i < n; i += gridDim.x * blockDim.x)
    m = fmaxf(m, x[i]);
  sm[threadIdx.x] = m; __syncthreads();
  for (int st = 128; st > 0; st >>= 1) {
    if ((int)threadIdx.x < st) sm[threadIdx.x] = fmaxf(sm[threadIdx.x], sm[threadIdx.x + st]);
    __syncthreads();
  }
  if (threadIdx.x == 0) part[blockIdx.x] = sm[0];
}
__global__ void reduce_max_final(const float* __restrict__ part, int n,
                                 float* __restrict__ outv) {
  __shared__ float sm[256];
  float m = -3.0e38f;
  for (int i = threadIdx.x; i < n; i += blockDim.x) m = fmaxf(m, part[i]);
  sm[threadIdx.x] = m; __syncthreads();
  for (int st = 128; st > 0; st >>= 1) {
    if ((int)threadIdx.x < st) sm[threadIdx.x] = fmaxf(sm[threadIdx.x], sm[threadIdx.x + st]);
    __syncthreads();
  }
  if (threadIdx.x == 0) outv[0] = sm[0];
}
__global__ void reduce_sumexp_partial(const float* __restrict__ x, int n,
                                      const float* __restrict__ gmax,
                                      float* __restrict__ part) {
  __shared__ float sm[256];
  const float gm = *gmax;
  float sa = 0.f;
  for (int i = blockIdx.x * blockDim.x + threadIdx.x; i < n; i += gridDim.x * blockDim.x)
    sa += __expf(x[i] - gm);
  sm[threadIdx.x] = sa; __syncthreads();
  for (int st = 128; st > 0; st >>= 1) {
    if ((int)threadIdx.x < st) sm[threadIdx.x] += sm[threadIdx.x + st];
    __syncthreads();
  }
  if (threadIdx.x == 0) part[blockIdx.x] = sm[0];
}
__global__ void reduce_suminv_final(const float* __restrict__ part, int n,
                                    float* __restrict__ outv) {
  __shared__ float sm[256];
  float sa = 0.f;
  for (int i = threadIdx.x; i < n; i += blockDim.x) sa += part[i];
  sm[threadIdx.x] = sa; __syncthreads();
  for (int st = 128; st > 0; st >>= 1) {
    if ((int)threadIdx.x < st) sm[threadIdx.x] += sm[threadIdx.x + st];
    __syncthreads();
  }
  if (threadIdx.x == 0) outv[0] = 1.0f / sm[0];
}

// ---------------- scatter (segment sums) + attn output ----------------
__global__ __launch_bounds__(256) void scatter_kernel(
    const float* __restrict__ logits, const float* __restrict__ msg,
    const float* __restrict__ cwraw, const float* __restrict__ relunit,
    const int* __restrict__ dstI, const float* __restrict__ gmax,
    const float* __restrict__ ginv, float* __restrict__ attn_out,
    float* __restrict__ agg, float* __restrict__ csum, int E_) {
  int e = blockIdx.x * blockDim.x + threadIdx.x;
  if (e >= E_) return;
  float a = __expf(logits[e] - *gmax) * (*ginv);
  attn_out[e] = a;
  int d = dstI[e];
  const float* mr = msg + (size_t)e * 64;
  float* ar = agg + (size_t)d * 64;
#pragma unroll 8
  for (int f = 0; f < 64; ++f) atomicAdd(ar + f, mr[f] * a);
  float cw = cwraw[e] * a;
#pragma unroll
  for (int c3 = 0; c3 < 3; ++c3)
    atomicAdd(csum + (size_t)d * 3 + c3, cw * relunit[(size_t)e * 3 + c3]);
}

__global__ void final_feats_kernel(const float* __restrict__ h,
                                   const float* __restrict__ agg,
                                   float* __restrict__ outp, long n) {
  long i = (long)blockIdx.x * blockDim.x + threadIdx.x;
  if (i < n) outp[i] = h[i] + agg[i];
}
__global__ void final_coords_kernel(const float* __restrict__ coords,
                                    const float* __restrict__ csum,
                                    float* __restrict__ outp, long n) {
  long i = (long)blockIdx.x * blockDim.x + threadIdx.x;
  if (i < n) outp[i] = coords[i] + 0.1f * csum[i];
}

// ---------------- host ----------------
extern "C" void kernel_launch(void* const* d_in, const int* in_sizes, int n_in,
                              void* d_out, int out_size, void* d_ws, size_t ws_size,
                              hipStream_t stream) {
  const float* node_feats = (const float*)d_in[0];
  const float* edge_feats = (const float*)d_in[1];
  const float* coords     = (const float*)d_in[2];
  const int*   srcI       = (const int*)d_in[3];
  const int*   dstI       = (const int*)d_in[4];
  const float* Wn    = (const float*)d_in[5];
  const float* bn    = (const float*)d_in[6];
  const float* cm_w1 = (const float*)d_in[7];
  const float* cm_b1 = (const float*)d_in[8];
  const float* cm_w2 = (const float*)d_in[9];
  const float* cm_b2 = (const float*)d_in[10];
  const float* de_w1 = (const float*)d_in[11];
  const float* de_b1 = (const float*)d_in[12];
  const float* de_w2 = (const float*)d_in[13];
  const float* de_b2 = (const float*)d_in[14];
  const float* am_w1 = (const float*)d_in[15];
  const float* am_b1 = (const float*)d_in[16];
  const float* am_w2 = (const float*)d_in[17];
  const float* am_b2 = (const float*)d_in[18];
  const float* mm_w1 = (const float*)d_in[19];
  const float* mm_b1 = (const float*)d_in[20];
  const float* mm_w2 = (const float*)d_in[21];
  const float* mm_b2 = (const float*)d_in[22];
  const float* cu_w1 = (const float*)d_in[23];
  const float* cu_b1 = (const float*)d_in[24];
  const float* cu_w2 = (const float*)d_in[25];
  const float* cu_b2 = (const float*)d_in[26];

  const int Nn = in_sizes[0] / 64;
  const int E_ = in_sizes[3];

  char* base = (char*)d_ws;
  size_t off = 0;
  auto ALLOC = [&](size_t b) -> void* {
    void* p = base + off;
    off = (off + b + 255) & ~(size_t)255;
    return p;
  };
  float*  h      = (float*) ALLOC((size_t)Nn * 64 * 4);
  __bf16* hbf    = (__bf16*)ALLOC((size_t)Nn * 64 * 2);
  __bf16* WnT    = (__bf16*)ALLOC((size_t)64 * 64 * 2);
  __bf16* amw1T  = (__bf16*)ALLOC((size_t)64 * 224 * 2);
  __bf16* mmw1T  = (__bf16*)ALLOC((size_t)128 * 224 * 2);
  __bf16* mmw2T  = (__bf16*)ALLOC((size_t)64 * 128 * 2);
  __bf16* cuw1T  = (__bf16*)ALLOC((size_t)64 * 64 * 2);
  float*  logits = (float*) ALLOC((size_t)E_ * 4);
  float*  msg    = (float*) ALLOC((size_t)E_ * 64 * 4);
  float*  cwraw  = (float*) ALLOC((size_t)E_ * 4);
  float*  relun  = (float*) ALLOC((size_t)E_ * 3 * 4);
  float*  agg    = (float*) ALLOC((size_t)Nn * 64 * 4);
  float*  csum   = (float*) ALLOC((size_t)Nn * 3 * 4);
  float*  pmax   = (float*) ALLOC(512 * 4);
  float*  psum   = (float*) ALLOC(512 * 4);
  float*  scal   = (float*) ALLOC(16);  // [0]=gmax, [1]=1/sum

  float* out_feats  = (float*)d_out;
  float* out_attn   = out_feats + (size_t)Nn * 64;
  float* out_coords = out_attn + E_;

  const long nf64 = (long)Nn * 64;
  const long nc3  = (long)Nn * 3;

  // zero accumulators (ws is poisoned, not re-zeroed between replays)
  fill_zero_kernel<<<(int)((nf64 + 255) / 256), 256, 0, stream>>>(agg, nf64);
  fill_zero_kernel<<<(int)((nc3 + 255) / 256), 256, 0, stream>>>(csum, nc3);

  // bf16 transposed weights (Wt[n][k]) for WMMA B fragments
  transpose_bf16_kernel<<<(64 * 64 + 255) / 256, 256, 0, stream>>>(Wn, WnT, 64, 64);
  transpose_bf16_kernel<<<(224 * 64 + 255) / 256, 256, 0, stream>>>(am_w1, amw1T, 224, 64);
  transpose_bf16_kernel<<<(224 * 128 + 255) / 256, 256, 0, stream>>>(mm_w1, mmw1T, 224, 128);
  transpose_bf16_kernel<<<(128 * 64 + 255) / 256, 256, 0, stream>>>(mm_w2, mmw2T, 128, 64);
  transpose_bf16_kernel<<<(64 * 64 + 255) / 256, 256, 0, stream>>>(cu_w1, cuw1T, 64, 64);

  // node transform
  const int ntiles = (Nn + 15) / 16;
  node_kernel<<<(ntiles + 7) / 8, 256, 0, stream>>>(node_feats, WnT, bn, h, hbf, Nn);

  // fused edge pipeline
  const int etiles = (E_ + 15) / 16;
  edge_kernel<<<(etiles + EK_WAVES - 1) / EK_WAVES, 128, 0, stream>>>(
      edge_feats, coords, srcI, dstI,
      cm_w1, cm_b1, cm_w2, cm_b2, de_w1, de_b1, de_w2, de_b2,
      hbf, amw1T, am_b1, am_w2, am_b2, mmw1T, mm_b1, mmw2T, mm_b2,
      cuw1T, cu_b1, cu_w2, cu_b2,
      logits, msg, cwraw, relun, E_);

  // softmax over all edges
  reduce_max_partial<<<512, 256, 0, stream>>>(logits, E_, pmax);
  reduce_max_final<<<1, 256, 0, stream>>>(pmax, 512, scal);
  reduce_sumexp_partial<<<512, 256, 0, stream>>>(logits, E_, scal, psum);
  reduce_suminv_final<<<1, 256, 0, stream>>>(psum, 512, scal + 1);

  // weighted scatter + attn output
  scatter_kernel<<<(E_ + 255) / 256, 256, 0, stream>>>(
      logits, msg, cwraw, relun, dstI, scal, scal + 1, out_attn, agg, csum, E_);

  // residual + coord update
  final_feats_kernel<<<(int)((nf64 + 255) / 256), 256, 0, stream>>>(h, agg, out_feats, nf64);
  final_coords_kernel<<<(int)((nc3 + 255) / 256), 256, 0, stream>>>(coords, csum, out_coords, nc3);
}
